// VectorQuantizer_12824772346353
// MI455X (gfx1250) — matrix operand order repaired and checked by reading, not compiled, not run on hardware
//
#include <hip/hip_runtime.h>
#include <hip/hip_bf16.h>
#include <math.h>

// ---------------- problem constants ----------------
#define NUM_EMB     1024
#define EMB_DIM     256
#define HW_TOT      1024            // 32*32
#define N_ROWS      65536           // 64*32*32
#define KCHUNK      128             // codebook entries staged in LDS per pass
#define ROWS_PER_WG 128             // 8 waves * 16 rows
#define NUM_WG      (N_ROWS / ROWS_PER_WG)  // 512

// d_out is the tuple flattened: z_q_st [N*D] fp32, loss, perplexity, idx [N]
#define OUT_ZQ_OFF   0
#define OUT_LOSS_OFF (N_ROWS * EMB_DIM)     // 16777216
#define OUT_PERP_OFF (OUT_LOSS_OFF + 1)
#define OUT_IDX_OFF  (OUT_LOSS_OFF + 2)

// ---------------- workspace layout (bytes) ----------------
#define WS_EMB_BF    0                       // u16[1024*256] = 512 KB
#define WS_EMB_NORM  524288                  // f32[1024]
#define WS_COUNTS    528384                  // i32[1024]
#define WS_PARTIALS  532480                  // f32[NUM_WG]

typedef __attribute__((ext_vector_type(16))) __bf16 bf16x16;
typedef __attribute__((ext_vector_type(8)))  float  f32x8;

static __device__ __forceinline__ unsigned pack2_bf16(float a, float b) {
  // round-to-nearest-even fp32 -> bf16, packed {hi=b, lo=a}
  unsigned ua = __float_as_uint(a);
  unsigned ub = __float_as_uint(b);
  unsigned ra = (ua + 0x7FFFu + ((ua >> 16) & 1u)) >> 16;
  unsigned rb = (ub + 0x7FFFu + ((ub >> 16) & 1u)) >> 16;
  return (rb << 16) | (ra & 0xFFFFu);
}

// ---------------------------------------------------------------------------
// Kernel 1: codebook prep. Block k handles entry k: convert to bf16, compute
// ||e_k||^2, zero the bincount slot.
// ---------------------------------------------------------------------------
__global__ __launch_bounds__(256) void vq_prep_kernel(
    const float* __restrict__ emb, unsigned short* __restrict__ emb_bf,
    float* __restrict__ emb_norm, int* __restrict__ counts)
{
  __shared__ float red[256];
  const int k = blockIdx.x;
  const int d = threadIdx.x;
  float v = emb[(size_t)k * EMB_DIM + d];
  unsigned u = __float_as_uint(v);
  emb_bf[(size_t)k * EMB_DIM + d] =
      (unsigned short)((u + 0x7FFFu + ((u >> 16) & 1u)) >> 16);
  red[d] = v * v;
  __syncthreads();
  for (int s = 128; s > 0; s >>= 1) {
    if (d < s) red[d] += red[d + s];
    __syncthreads();
  }
  if (d == 0) {
    emb_norm[k] = red[0];
    counts[k] = 0;
  }
}

// ---------------------------------------------------------------------------
// Kernel 2: fused GEMM(bf16 WMMA) + argmin + gather + loss partial.
// Workgroup = 256 threads = 8 waves; wave w owns rows [wg_base+16w, +16).
// A tile (16 rows x 256 dims, bf16) lives in VGPRs for the whole K sweep.
// Codebook streamed through LDS in 128-entry chunks via async-to-LDS DMA.
// ---------------------------------------------------------------------------
__global__ __launch_bounds__(256) void vq_main_kernel(
    const float* __restrict__ z_e, const float* __restrict__ emb,
    const unsigned short* __restrict__ emb_bf,
    const float* __restrict__ emb_norm,
    int* __restrict__ counts, float* __restrict__ partials,
    float* __restrict__ out)
{
  __shared__ alignas(16) unsigned short lds_emb[KCHUNK * EMB_DIM]; // 64 KB
  __shared__ float lds_ebn[KCHUNK];
  __shared__ int   lds_idx[ROWS_PER_WG];
  __shared__ float lds_red[256];

  const int tid  = threadIdx.x;
  const int wave = tid >> 5;
  const int lane = tid & 31;
  const int m16  = lane & 15;      // M (rows) for A lanes, N (entry) for B/C lanes
  const int half = lane >> 4;
  const int wg_rowbase = blockIdx.x * ROWS_PER_WG;
  const int rowbase    = wg_rowbase + wave * 16;

  // ---- load A tile: row n = rowbase + m16; x[n][d] = z_e[b][d][hw] ----
  // A 16x32 layout (contiguous k, matching B): lanes 0-15 M=0..15; VGPR v in 0..7 -> K = half*16 + 2v.
  union AF { bf16x16 v; unsigned u[8]; };
  AF a[8];
  {
    const int n  = rowbase + m16;
    const int bb = n >> 10;
    const int hw = n & (HW_TOT - 1);
    const float* zrow = z_e + (size_t)bb * EMB_DIM * HW_TOT + hw; // dim d at zrow[d*HW]
#pragma unroll
    for (int c = 0; c < 8; ++c) {
#pragma unroll
      for (int v = 0; v < 8; ++v) {
        const int k0 = c * 32 + half * 16 + 2 * v;
        const float f0 = zrow[(size_t)k0 * HW_TOT];
        const float f1 = zrow[(size_t)(k0 + 1) * HW_TOT];
        a[c].u[v] = pack2_bf16(f0, f1);
      }
    }
  }

  // running argmin of (||e||^2 - 2 x.e)  (||x||^2 is row-constant -> irrelevant)
  float best[8];
  int   bidx[8];
#pragma unroll
  for (int r = 0; r < 8; ++r) { best[r] = 3.0e38f; bidx[r] = 0; }

  // LDS byte offset of the staging buffer (generic LDS address keeps the
  // wave-relative offset in the low 32 bits).
  const unsigned lds_base = (unsigned)(uintptr_t)(&lds_emb[0]);

  for (int kc = 0; kc < NUM_EMB / KCHUNK; ++kc) {
    __syncthreads();   // previous chunk fully consumed by all waves
    // ---- async DMA stage of the 64 KB codebook chunk into LDS ----
    // 256 threads x 16 x b128 = 64 KB, VGPR-bypassing (ASYNCcnt-tracked).
    {
      const char* gsrc = (const char*)(emb_bf + (size_t)kc * KCHUNK * EMB_DIM);
#pragma unroll
      for (int i = 0; i < 16; ++i) {
        const unsigned          loff = lds_base + (unsigned)(tid + 256 * i) * 16u;
        const unsigned long long gaddr =
            (unsigned long long)(uintptr_t)(gsrc + (size_t)(tid + 256 * i) * 16u);
        asm volatile("global_load_async_to_lds_b128 %0, %1, off"
                     :: "v"(loff), "v"(gaddr) : "memory");
      }
      if (tid < KCHUNK) lds_ebn[tid] = emb_norm[kc * KCHUNK + tid];
      asm volatile("s_wait_asynccnt 0x0" ::: "memory");
    }
    __syncthreads();

    // sweep the 128 staged entries in 16-wide N tiles
    for (int nc = 0; nc < KCHUNK / 16; ++nc) {
      const int e = nc * 16 + m16;                       // entry index in chunk
      const float ebn = lds_ebn[e];
      // B 32x16 as used here (contiguous k, matching A): VGPR v holds K = 2v,2v+1 (+16 for lanes 16-31),
      // N = lane&15  -> lane reads d = dc*32 + half*16 + [0..15] of entry e.
      const unsigned short* brow = lds_emb + (size_t)e * EMB_DIM + half * 16;
      // hoist ALL B fragments (16 x ds_load_b128) ahead of the WMMA chain so
      // the scheduler can issue partial s_wait_dscnt and overlap LDS latency.
      uint4 bq[16];
#pragma unroll
      for (int dc = 0; dc < 8; ++dc) {
        const uint4* p = (const uint4*)(brow + dc * 32);
        bq[2 * dc]     = p[0];
        bq[2 * dc + 1] = p[1];
      }
      f32x8 cacc = {0.f, 0.f, 0.f, 0.f, 0.f, 0.f, 0.f, 0.f};
#pragma unroll
      for (int dc = 0; dc < 8; ++dc) {
        union BFr { bf16x16 v; uint4 q[2]; } bfr;
        bfr.q[0] = bq[2 * dc];
        bfr.q[1] = bq[2 * dc + 1];
        cacc = __builtin_amdgcn_wmma_f32_16x16x32_bf16(
            false, a[dc].v, false, bfr.v, (short)0, cacc, false, false);
      }
      const int gidx = kc * KCHUNK + nc * 16 + m16;
#pragma unroll
      for (int r = 0; r < 8; ++r) {
        // C layout: lane<16 -> (M=r, N=lane); lane>=16 -> (M=r+8, N=lane-16)
        const float dist = ebn - 2.0f * cacc[r];
        if (dist < best[r]) { best[r] = dist; bidx[r] = gidx; }
      }
    }
  }

  // argmin across the 16 N-lanes (halves are distinct M rows: xor<16 stays in half)
#pragma unroll
  for (int off = 8; off >= 1; off >>= 1) {
#pragma unroll
    for (int r = 0; r < 8; ++r) {
      const float ob = __shfl_xor(best[r], off, 32);
      const int   oi = __shfl_xor(bidx[r], off, 32);
      if (ob < best[r]) { best[r] = ob; bidx[r] = oi; }
    }
  }
  if (m16 == 0) {
#pragma unroll
    for (int r = 0; r < 8; ++r) {
      const int m   = r + half * 8;
      const int row = rowbase + m;
      const int id  = bidx[r];
      lds_idx[wave * 16 + m] = id;
      out[(size_t)OUT_IDX_OFF + row] = (float)id;
      atomicAdd(&counts[id], 1);
    }
  }
  __syncthreads();

  // ---- fused gather (z_q in NCHW) + loss partial: mean((z_q - z)^2) ----
  {
    float lsum = 0.f;
    const int rr = tid & 127;        // row within WG
    const int dg = tid >> 7;         // which half of the channel dim
    const int grow = wg_rowbase + rr;
    const int gb = grow >> 10;
    const int ghw = grow & (HW_TOT - 1);
    const int gid = lds_idx[rr];
    const float* erow = emb + (size_t)gid * EMB_DIM;
    const float* zr   = z_e + (size_t)gb * EMB_DIM * HW_TOT + ghw;
    float*       orow = out + (size_t)gb * EMB_DIM * HW_TOT + ghw;
#pragma unroll 4
    for (int dd = 0; dd < 128; ++dd) {
      const int d = dg * 128 + dd;
      const float ev = erow[d];
      const float zv = zr[(size_t)d * HW_TOT];
      orow[(size_t)d * HW_TOT] = ev;           // z_q_st == z_q numerically
      const float df = ev - zv;
      lsum += df * df;
    }
    lds_red[tid] = lsum;
    __syncthreads();
    for (int s = 128; s > 0; s >>= 1) {
      if (tid < s) lds_red[tid] += lds_red[tid + s];
      __syncthreads();
    }
    if (tid == 0) partials[blockIdx.x] = lds_red[0];   // deterministic partials
  }
}

// ---------------------------------------------------------------------------
// Kernel 3: finalize loss (= 1.25 * mse) and perplexity.
// ---------------------------------------------------------------------------
__global__ __launch_bounds__(256) void vq_final_kernel(
    const int* __restrict__ counts, const float* __restrict__ partials,
    float* __restrict__ out)
{
  __shared__ float redH[256];
  __shared__ float redL[256];
  const int t = threadIdx.x;
  float h = 0.f;
  for (int i = t; i < NUM_EMB; i += 256) {
    const float p = (float)counts[i] * (1.0f / (float)N_ROWS);
    h += p * logf(p + 1e-10f);
  }
  float l = 0.f;
  for (int i = t; i < NUM_WG; i += 256) l += partials[i];
  redH[t] = h; redL[t] = l;
  __syncthreads();
  for (int s = 128; s > 0; s >>= 1) {
    if (t < s) { redH[t] += redH[t + s]; redL[t] += redL[t + s]; }
    __syncthreads();
  }
  if (t == 0) {
    out[OUT_PERP_OFF] = expf(-redH[0]);
    out[OUT_LOSS_OFF] = redL[0] * 1.25f / (float)((size_t)N_ROWS * EMB_DIM);
  }
}

// ---------------------------------------------------------------------------
extern "C" void kernel_launch(void* const* d_in, const int* in_sizes, int n_in,
                              void* d_out, int out_size, void* d_ws, size_t ws_size,
                              hipStream_t stream) {
  const float* z_e = (const float*)d_in[0];
  const float* emb = (const float*)d_in[1];
  float* out = (float*)d_out;
  char* ws = (char*)d_ws;

  unsigned short* emb_bf  = (unsigned short*)(ws + WS_EMB_BF);
  float*          embnorm = (float*)(ws + WS_EMB_NORM);
  int*            counts  = (int*)(ws + WS_COUNTS);
  float*          parts   = (float*)(ws + WS_PARTIALS);

  vq_prep_kernel<<<NUM_EMB, 256, 0, stream>>>(emb, emb_bf, embnorm, counts);
  vq_main_kernel<<<NUM_WG, 256, 0, stream>>>(z_e, emb, emb_bf, embnorm,
                                             counts, parts, out);
  vq_final_kernel<<<1, 256, 0, stream>>>(counts, parts, out);
}
